// BasicCGInducer_65618510348901
// MI455X (gfx1250) — matrix operand order repaired and checked by reading, not compiled, not run on hardware
//
#include <hip/hip_runtime.h>
#include <hip/hip_bf16.h>

typedef _Float16 v16h __attribute__((ext_vector_type(16)));
typedef _Float16 v8h  __attribute__((ext_vector_type(8)));
typedef float    v8f  __attribute__((ext_vector_type(8)));

#define NEGF (-1e9f)
#define MINIT (-1e30f)

// D=256, V=10000, B=16, N=22, CF=400, CR=120, CA=48

// ---------------------------------------------------------------- async global->LDS helpers (CDNA5)
__device__ __forceinline__ void async_copy_b128(void* lds_ptr, const float* gptr) {
    unsigned lds_off = (unsigned)(unsigned long long)lds_ptr;   // low 32 bits = LDS offset
    asm volatile("global_load_async_to_lds_b128 %0, %1, off"
                 :: "v"(lds_off), "v"(gptr) : "memory");
}
__device__ __forceinline__ void wait_async0() {
    asm volatile("s_wait_asynccnt 0x0" ::: "memory");
}

// ---------------------------------------------------------------- beta init
__global__ void init_beta_kernel(float* beta, unsigned long long n) {
    unsigned long long i = (unsigned long long)blockIdx.x * blockDim.x + threadIdx.x;
    if (i < n) beta[i] = NEGF;
}

// ---------------------------------------------------------------- weight pack: Wt[n][k] = (f16)W[k][n]
__global__ void pack_w_kernel(const float* __restrict__ W, _Float16* __restrict__ Wt,
                              int K, int N) {
    long long idx = (long long)blockIdx.x * blockDim.x + threadIdx.x;
    if (idx >= (long long)K * N) return;
    int n = (int)(idx / K), k = (int)(idx % K);
    Wt[(size_t)n * K + k] = (_Float16)W[(size_t)k * N + n];
}

// ---------------------------------------------------------------- root log-softmax
__global__ void root_kernel(const float* root_w, const float* root_b,
                            const float* root_mask, float* root_scores) {
    __shared__ float sm[512];
    int t = threadIdx.x;
    float x = (t < 400) ? (root_mask[t] + root_w[t] + root_b[t]) : MINIT;
    sm[t] = x; __syncthreads();
    for (int o = 256; o > 0; o >>= 1) { if (t < o) sm[t] = fmaxf(sm[t], sm[t + o]); __syncthreads(); }
    float mx = sm[0]; __syncthreads();
    sm[t] = (t < 400) ? __expf(x - mx) : 0.f; __syncthreads();
    for (int o = 256; o > 0; o >>= 1) { if (t < o) sm[t] += sm[t + o]; __syncthreads(); }
    float lse = mx + __logf(sm[0]);
    if (t < 400) root_scores[t] = x - lse;
}

// ---------------------------------------------------------------- WMMA fragment helpers
// A fragment (16x32, K=256 LDS tile): two aligned v8h LDS reads (ISA 7.12.2 A layout)
__device__ __forceinline__ v16h a_frag_lds(const _Float16* As, int lane, int k0) {
    int m = lane & 15, half = (lane >> 4) & 1;
    const _Float16* ap = As + m * 256;
    v8h lo = *(const v8h*)(ap + k0 + half * 8);
    v8h hi = *(const v8h*)(ap + k0 + 16 + half * 8);
    return __builtin_shufflevector(lo, hi, 0,1,2,3,4,5,6,7,8,9,10,11,12,13,14,15);
}
// B fragment (32x16) from packed-transposed f16 weights Wt[n][k]: two global b128 per lane
__device__ __forceinline__ v16h b_frag_packed(const _Float16* Wt, int K, int col0,
                                              int k0, int lane) {
    int n = lane & 15, half = (lane >> 4) & 1;
    const _Float16* wp = Wt + (size_t)(col0 + n) * K + k0 + half * 8;
    v8h lo = *(const v8h*)(wp);
    v8h hi = *(const v8h*)(wp + 16);
    return __builtin_shufflevector(lo, hi, 0,1,2,3,4,5,6,7,8,9,10,11,12,13,14,15);
}

// ---------------------------------------------------------------- MLP GEMM: Out = f(A@W + bias) [+ skip]
// M=400, K=256, N=256. grid(25,2), block 256 (8 waves); A tile staged once in LDS.
template <int RELU, int SKIP>
__global__ void gemm_mlp_kernel(const float* __restrict__ A, const _Float16* __restrict__ Wt,
                                const float* __restrict__ bias, const float* __restrict__ skip,
                                float* __restrict__ Out) {
    __shared__ __align__(16) _Float16 As[16 * 256];
    int t = threadIdx.x, lane = t & 31, wave = t >> 5;
    int row0 = blockIdx.x * 16;
    for (int i = t; i < 16 * 256; i += 256)
        As[i] = (_Float16)A[(size_t)(row0 + (i >> 8)) * 256 + (i & 255)];
    __syncthreads();

    int col0 = (blockIdx.y * 8 + wave) * 16;
    v8f acc = {};
#pragma unroll
    for (int k0 = 0; k0 < 256; k0 += 32) {
        v16h a = a_frag_lds(As, lane, k0);
        v16h b = b_frag_packed(Wt, 256, col0, k0, lane);
        acc = __builtin_amdgcn_wmma_f32_16x16x32_f16(false, a, false, b,
                                                     (short)0, acc, false, false);
    }
    int n = lane & 15, half = lane >> 4;
    int col = col0 + n;
    float bcol = bias[col];
#pragma unroll
    for (int r = 0; r < 8; ++r) {
        int row = row0 + half * 8 + r;
        float v = acc[r] + bcol;
        if (RELU) v = fmaxf(v, 0.f);
        if (SKIP) v += skip[(size_t)row * 256 + col];
        Out[(size_t)row * 256 + col] = v;
    }
}

// ---------------------------------------------------------------- split log-softmax (400 x 2)
__global__ void split_kernel(const float* h, const float* Wo, const float* bo,
                             float* split_scores) {
    int f = blockIdx.x * blockDim.x + threadIdx.x;
    if (f >= 400) return;
    const float* hr = h + (size_t)f * 256;
    float s0 = bo[0], s1 = bo[1];
    for (int k = 0; k < 256; ++k) { float hv = hr[k]; s0 += hv * Wo[k * 2]; s1 += hv * Wo[k * 2 + 1]; }
    float mx = fmaxf(s0, s1);
    float lse = mx + __logf(__expf(s0 - mx) + __expf(s1 - mx));
    split_scores[f * 2]     = s0 - lse;
    split_scores[f * 2 + 1] = s1 - lse;
}

// ---------------------------------------------------------------- rule log-softmax -> gL,gR
__global__ void rule_kernel(const float* rule_W, const float* rule_b,
                            const float* split_scores, float* gL, float* gR) {
    __shared__ float sm[128];
    int r = blockIdx.x;           // 0..119
    int j = threadIdx.x;          // 128 threads, 96 active
    float x = (j < 96) ? (rule_W[j * 120 + r] + rule_b[j]) : MINIT;
    sm[j] = x; __syncthreads();
    for (int o = 64; o > 0; o >>= 1) { if (j < o) sm[j] = fmaxf(sm[j], sm[j + o]); __syncthreads(); }
    float mx = sm[0]; __syncthreads();
    sm[j] = (j < 96) ? __expf(x - mx) : 0.f; __syncthreads();
    for (int o = 64; o > 0; o >>= 1) { if (j < o) sm[j] += sm[j + o]; __syncthreads(); }
    float lse = mx + __logf(sm[0]);
    float ss0 = split_scores[r * 2];
    if (j < 48)       gL[r * 48 + j]        = (x - lse) + ss0;
    else if (j < 96)  gR[r * 48 + (j - 48)] = (x - lse) + ss0;
}

// ---------------------------------------------------------------- emission denom: fused GEMM + streaming LSE
// 25 row-tile blocks x 256 threads (8 waves); waves stream 625 column tiles of packed emit_W.
__global__ void emit_denom_kernel(const float* __restrict__ nt_emb,
                                  const _Float16* __restrict__ emit_Wt,   // [V][256] f16
                                  const float* __restrict__ emit_b, float* denom) {
    __shared__ __align__(16) float smem[4096];     // 16 KB: A-tile (f16) then (m,s) partials
    _Float16* As = (_Float16*)smem;                // 16 x 256 halves
    int t = threadIdx.x, lane = t & 31, wave = t >> 5;
    int row0 = blockIdx.x * 16;
    for (int i = t; i < 16 * 256; i += 256)
        As[i] = (_Float16)nt_emb[(size_t)(row0 + (i >> 8)) * 256 + (i & 255)];
    __syncthreads();

    int n = lane & 15, half = lane >> 4;
    float m[8], s[8];
#pragma unroll
    for (int r = 0; r < 8; ++r) { m[r] = MINIT; s[r] = 0.f; }

    for (int jt = wave; jt < 625; jt += 8) {       // 10000/16 column tiles
        int col0 = jt * 16;
        v8f acc = {};
#pragma unroll
        for (int k0 = 0; k0 < 256; k0 += 32) {
            v16h a = a_frag_lds(As, lane, k0);
            v16h b = b_frag_packed(emit_Wt, 256, col0, k0, lane);
            acc = __builtin_amdgcn_wmma_f32_16x16x32_f16(false, a, false, b,
                                                         (short)0, acc, false, false);
        }
        float eb = emit_b[col0 + n];
#pragma unroll
        for (int r = 0; r < 8; ++r) {              // online logsumexp
            float x = acc[r] + eb;
            float mn = fmaxf(m[r], x);
            s[r] = s[r] * __expf(m[r] - mn) + __expf(x - mn);
            m[r] = mn;
        }
    }
    __syncthreads();                               // A-tile dead; reuse smem for partials
    int slot = wave * 16 + n;                      // 128 partials per row
#pragma unroll
    for (int r = 0; r < 8; ++r) {
        int row = half * 8 + r;
        smem[row * 256 + slot * 2]     = m[r];
        smem[row * 256 + slot * 2 + 1] = s[r];
    }
    __syncthreads();
    if (t < 16) {
        float M = MINIT, S = 0.f;
        for (int sl = 0; sl < 128; ++sl) {
            float mi = smem[t * 256 + sl * 2], si = smem[t * 256 + sl * 2 + 1];
            float mn = fmaxf(M, mi);
            S = S * __expf(M - mn) + si * __expf(mi - mn);
            M = mn;
        }
        denom[row0 + t] = M + __logf(S);
    }
}

// ---------------------------------------------------------------- terminal scores -> beta diagonal
__global__ void term_kernel(const float* nt_emb, const float* emit_W, const float* emit_b,
                            const int* words, const float* split_scores,
                            const float* denom, float* beta) {
    int bn = blockIdx.x;                 // b*N + n  (352 blocks)
    int b = bn / 22, nn = bn % 22;
    int f = threadIdx.x;
    if (f >= 400) return;
    int w = words[bn];
    const float* arow = nt_emb + (size_t)f * 256;
    float dot = 0.f;
    for (int k = 0; k < 256; ++k) dot += arow[k] * emit_W[(size_t)k * 10000 + w];
    float val = split_scores[f * 2 + 1] + (dot + emit_b[w] - denom[f]);
    beta[((((size_t)nn * 22) + nn) * 16 + b) * 400 + f] = val;
}

// ---------------------------------------------------------------- one CKY width step
// Span vectors staged global->LDS with async b128 copies (ASYNCcnt path).
__global__ void cky_kernel(float* beta, const float* __restrict__ gL,
                           const float* __restrict__ gR, const int* __restrict__ lf,
                           const int* __restrict__ rf, int w) {
    __shared__ __align__(16) float Lv[400];
    __shared__ __align__(16) float Rv[400];
    int st = blockIdx.x, b = blockIdx.y;
    int t = threadIdx.x;                 // 128 threads, 120 active for compute
    float mL = MINIT, sL = 0.f, mR = MINIT, sR = 0.f;
    for (int k = 1; k < w; ++k) {
        const float* lptr = beta + ((((size_t)st * 22) + (st + k - 1)) * 16 + b) * 400;
        const float* rptr = beta + ((((size_t)(st + k) * 22) + (st + w - 1)) * 16 + b) * 400;
        if (t < 100) {                   // 100 x 16B = 400 floats each
            async_copy_b128(Lv + 4 * t, lptr + 4 * t);
            async_copy_b128(Rv + 4 * t, rptr + 4 * t);
        }
        wait_async0();
        __syncthreads();
        if (t < 120) {
            int r = t;
#pragma unroll 4
            for (int a = 0; a < 48; ++a) {
                int ia = r * 48 + a;
                float xL = gL[ia] + Lv[a] + Rv[rf[ia]];
                float mn = fmaxf(mL, xL);
                sL = sL * __expf(mL - mn) + __expf(xL - mn);
                mL = mn;
                float xR = gR[ia] + Lv[lf[ia]] + Rv[a];
                mn = fmaxf(mR, xR);
                sR = sR * __expf(mR - mn) + __expf(xR - mn);
                mR = mn;
            }
        }
        __syncthreads();
    }
    if (t < 120) {
        float vL = mL + __logf(sL);
        float vR = mR + __logf(sR);
        float mm = fmaxf(vL, vR);
        float val = mm + __logf(__expf(vL - mm) + __expf(vR - mm));
        beta[((((size_t)st * 22) + (st + w - 1)) * 16 + b) * 400 + t] = val;
    }
}

// ---------------------------------------------------------------- final NLL
__global__ void final_kernel(const float* beta, const float* root_scores, float* out) {
    __shared__ float sm[64];
    int b = blockIdx.x, t = threadIdx.x;
    const float* bp = beta + (((size_t)0 * 22 + 21) * 16 + b) * 400;
    float mx = MINIT;
    for (int f = t; f < 400; f += 64) mx = fmaxf(mx, root_scores[f] + bp[f]);
    sm[t] = mx; __syncthreads();
    for (int o = 32; o > 0; o >>= 1) { if (t < o) sm[t] = fmaxf(sm[t], sm[t + o]); __syncthreads(); }
    mx = sm[0]; __syncthreads();
    float s = 0.f;
    for (int f = t; f < 400; f += 64) s += __expf(root_scores[f] + bp[f] - mx);
    sm[t] = s; __syncthreads();
    for (int o = 32; o > 0; o >>= 1) { if (t < o) sm[t] += sm[t + o]; __syncthreads(); }
    if (t == 0) out[b] = -(mx + __logf(sm[0]));
}

// ---------------------------------------------------------------- host launch
extern "C" void kernel_launch(void* const* d_in, const int* in_sizes, int n_in,
                              void* d_out, int out_size, void* d_ws, size_t ws_size,
                              hipStream_t stream) {
    (void)in_sizes; (void)n_in; (void)out_size; (void)ws_size;
    const float* nt_emb     = (const float*)d_in[1];   // d_in[0] fake_emb == identity, unused
    const float* rule_W     = (const float*)d_in[2];
    const float* rule_b     = (const float*)d_in[3];
    const float* root_w     = (const float*)d_in[4];
    const float* root_b     = (const float*)d_in[5];
    const float* root_mask  = (const float*)d_in[6];
    const float* split_W1   = (const float*)d_in[7];
    const float* split_b1   = (const float*)d_in[8];
    const float* res_W      = (const float*)d_in[9];
    const float* res_b      = (const float*)d_in[10];
    const float* split_Wo   = (const float*)d_in[11];
    const float* split_bo   = (const float*)d_in[12];
    const float* emit_W     = (const float*)d_in[13];
    const float* emit_b     = (const float*)d_in[14];
    const int*   words      = (const int*)d_in[15];
    const int*   lf         = (const int*)d_in[16];
    const int*   rf         = (const int*)d_in[17];
    float* out = (float*)d_out;

    float* ws = (float*)d_ws;
    size_t off = 0;
    float* beta         = ws + off; off += (size_t)22 * 22 * 16 * 400;  // 3,097,600
    float* hA           = ws + off; off += 400 * 256;
    float* hB           = ws + off; off += 400 * 256;
    float* tb           = ws + off; off += 400 * 256;
    float* root_scores  = ws + off; off += 400;
    float* split_scores = ws + off; off += 800;
    float* denom        = ws + off; off += 400;
    float* gLbuf        = ws + off; off += 120 * 48;
    float* gRbuf        = ws + off; off += 120 * 48;
    // f16 packed-weight region (off is a multiple of 4 -> 16B aligned)
    _Float16* h16 = (_Float16*)(ws + off);
    _Float16* W1p   = h16;                     // 256*256
    _Float16* Rp[4] = { h16 + 1 * 65536, h16 + 2 * 65536, h16 + 3 * 65536, h16 + 4 * 65536 };
    _Float16* Ep    = h16 + 5 * 65536;         // 10000*256

    const unsigned long long betaN = 3097600ull;
    init_beta_kernel<<<dim3((unsigned)((betaN + 255) / 256)), 256, 0, stream>>>(beta, betaN);
    root_kernel<<<1, 512, 0, stream>>>(root_w, root_b, root_mask, root_scores);

    // pack weights to transposed f16
    pack_w_kernel<<<256, 256, 0, stream>>>(split_W1, W1p, 256, 256);
    for (int i = 0; i < 4; ++i)
        pack_w_kernel<<<256, 256, 0, stream>>>(res_W + (size_t)i * 65536, Rp[i], 256, 256);
    pack_w_kernel<<<10000, 256, 0, stream>>>(emit_W, Ep, 256, 10000);

    dim3 gb(25, 2), bb(256);
    gemm_mlp_kernel<0, 0><<<gb, bb, 0, stream>>>(nt_emb, W1p,   split_b1,    nullptr, hA);
    gemm_mlp_kernel<1, 0><<<gb, bb, 0, stream>>>(hA,     Rp[0], res_b + 0,   nullptr, tb);
    gemm_mlp_kernel<1, 1><<<gb, bb, 0, stream>>>(tb,     Rp[1], res_b + 256, hA,      hB);
    gemm_mlp_kernel<1, 0><<<gb, bb, 0, stream>>>(hB,     Rp[2], res_b + 512, nullptr, tb);
    gemm_mlp_kernel<1, 1><<<gb, bb, 0, stream>>>(tb,     Rp[3], res_b + 768, hB,      hA);

    split_kernel<<<2, 256, 0, stream>>>(hA, split_Wo, split_bo, split_scores);
    rule_kernel<<<120, 128, 0, stream>>>(rule_W, rule_b, split_scores, gLbuf, gRbuf);

    emit_denom_kernel<<<25, 256, 0, stream>>>(nt_emb, Ep, emit_b, denom);
    term_kernel<<<16 * 22, 512, 0, stream>>>(nt_emb, emit_W, emit_b, words, split_scores, denom, beta);

    for (int w = 2; w <= 22; ++w) {
        dim3 g((unsigned)(22 - w + 1), 16);
        cky_kernel<<<g, 128, 0, stream>>>(beta, gLbuf, gRbuf, lf, rf, w);
    }
    final_kernel<<<16, 64, 0, stream>>>(beta, root_scores, out);
}
